// highwayNet_16999480557690
// MI455X (gfx1250) — matrix operand order repaired and checked by reading, not compiled, not run on hardware
//
#include <hip/hip_runtime.h>
#include <hip/hip_bf16.h>
#include <math.h>

// ---------------------------------------------------------------------------
// Model dims
// ---------------------------------------------------------------------------
#define T_IN   16
#define T_OUT  25
#define BATCH  4096
#define KNBR   32
#define NTOT   (BATCH * KNBR)     // 131072
#define ENC    64
#define DEC    128
#define DYN    32
#define EMB    32

typedef __attribute__((ext_vector_type(16))) _Float16 v16h;
typedef __attribute__((ext_vector_type(8)))  float    v8f;
typedef _Float16 half_t;

// ---------------------------------------------------------------------------
// WMMA helpers (CDNA5: v_wmma_f32_16x16x32_f16, wave32)
// ---------------------------------------------------------------------------
__device__ __forceinline__ v8f wmma16(v16h a, v16h b, v8f c) {
  // 8 args: (neg_a, A, neg_b, B, c_mod, C, reuse_a, reuse_b)
  return __builtin_amdgcn_wmma_f32_16x16x32_f16(false, a, false, b, (short)0, c,
                                                false, false);
}

__device__ __forceinline__ v8f v8f_zero() {
  v8f v;
#pragma unroll
  for (int i = 0; i < 8; ++i) v[i] = 0.f;
  return v;
}
__device__ __forceinline__ v16h v16h_zero() {
  v16h v;
#pragma unroll
  for (int i = 0; i < 16; ++i) v[i] = (_Float16)0;
  return v;
}

// A-matrix 16x32 f16, k order used here (not AMD's table order): lane(0..15)=M, lanes16-31 same M,
// VGPR g holds K pair K = khalf + 2g, khalf = (lane>=16) ? 16 : 0 (matches B).
// All k values are even -> contiguous pairs.
__device__ __forceinline__ int a_k_of_g(int g, int lane) {
  int khalf = ((lane >> 4) & 1) << 4;
  return khalf + 2 * g;
}

// Load an A fragment from a row-major f16 matrix (16 rows x >=k0+32 cols).
__device__ __forceinline__ v16h load_a_f16(const half_t* base, int ld, int k0,
                                           int lane) {
  int m = lane & 15;
  v16h a;
#pragma unroll
  for (int g = 0; g < 8; ++g) {
    int k = k0 + a_k_of_g(g, lane);
    a[2 * g]     = base[m * ld + k];
    a[2 * g + 1] = base[m * ld + k + 1];
  }
  return a;
}

// Load a B fragment (32x16, i.e. B[k][n]) from a row-major weight W[out,in]:
// B[k][n] = W[n0+n][k0+k].  k order used here (same as A, not AMD's table order): lanes 0-15 hold
// K=0..15 (contiguous 16 half elements), lanes 16-31 hold K=16..31.
// The 32 bytes are contiguous and 32B aligned (ld, k0 multiples of 16) ->
// single v16h dereference so the compiler emits b128 loads (global or ds).
__device__ __forceinline__ v16h load_b_w(const half_t* W, int ld, int n0,
                                         int k0, int lane) {
  int n  = lane & 15;
  int kb = k0 + (((lane >> 4) & 1) << 4);
  return *(const v16h*)(W + (n0 + n) * ld + kb);
}

// Cooperative global->LDS staging of a f16 weight matrix (16B chunks).
__device__ __forceinline__ void stage_lds(half_t* dst, const half_t* src,
                                          int halfs, int tid, int nthreads) {
  uint4* d = (uint4*)dst;
  const uint4* s = (const uint4*)src;
  int n = halfs >> 3;  // 8 halfs per 16B
  for (int i = tid; i < n; i += nthreads) d[i] = s[i];
}

__device__ __forceinline__ float sigf(float x) { return 1.f / (1.f + __expf(-x)); }
__device__ __forceinline__ float lrelu(float x) { return x >= 0.f ? x : 0.1f * x; }

// ---------------------------------------------------------------------------
// K0: f32 -> f16 weight conversion
// ---------------------------------------------------------------------------
__global__ void f32_to_f16_kernel(const float* __restrict__ src,
                                  half_t* __restrict__ dst, int n) {
  int i = blockIdx.x * blockDim.x + threadIdx.x;
  if (i < n) dst[i] = (half_t)src[i];
}

// ---------------------------------------------------------------------------
// K1: ego embedding  emb[t,b,e] = lrelu(hist[t,b,:]@Wip[e,:] + bip[e]), f16
// ---------------------------------------------------------------------------
__global__ void embed_kernel(const float* __restrict__ hist,
                             const float* __restrict__ Wip,
                             const float* __restrict__ bip,
                             half_t* __restrict__ emb) {
  int idx = blockIdx.x * blockDim.x + threadIdx.x;  // (t*B+b)*EMB + e
  const int TOT = T_IN * BATCH * EMB;
  if (idx >= TOT) return;
  int e   = idx & (EMB - 1);
  int row = idx >> 5;
  float x0 = hist[row * 2 + 0], x1 = hist[row * 2 + 1];
  float v  = x0 * Wip[e * 2 + 0] + x1 * Wip[e * 2 + 1] + bip[e];
  emb[idx] = (half_t)lrelu(v);
}

// ---------------------------------------------------------------------------
// K2: neighbor MLP + fused segment-max.  One wave per hero (32 neighbors =
// two 16-row M tiles).  A built on the fly: A[m][2t+d] = hist[t,b,d]-nbrs[t,n,d].
// max(rows) commutes with +bias and lrelu, so reduce first, then activate.
// ---------------------------------------------------------------------------
__global__ void neighbor_scene_kernel(const float* __restrict__ hist,
                                      const float* __restrict__ nbrs,
                                      const half_t* __restrict__ WnbrH,
                                      const float* __restrict__ bnbr,
                                      float* __restrict__ scene /*[B,ENC]*/) {
  __shared__ __align__(32) half_t sW[ENC * 2 * T_IN];  // 4 KB
  stage_lds(sW, WnbrH, ENC * 2 * T_IN, threadIdx.x, blockDim.x);
  __syncthreads();

  int lane = threadIdx.x & 31;
  int b    = blockIdx.x * (blockDim.x >> 5) + (threadIdx.x >> 5);  // hero id
  int m    = lane & 15;

  v16h aH[2];
#pragma unroll
  for (int h2 = 0; h2 < 2; ++h2) {
    int n = b * KNBR + h2 * 16 + m;
#pragma unroll
    for (int g = 0; g < 8; ++g) {
      int k = a_k_of_g(g, lane);            // even -> (t, d=0),(t, d=1)
      int t = k >> 1;
      float hx = hist[(t * BATCH + b) * 2 + 0];
      float hy = hist[(t * BATCH + b) * 2 + 1];
      float nx = nbrs[(t * NTOT + n) * 2 + 0];
      float ny = nbrs[(t * NTOT + n) * 2 + 1];
      aH[h2][2 * g]     = (half_t)(hx - nx);
      aH[h2][2 * g + 1] = (half_t)(hy - ny);
    }
  }

#pragma unroll
  for (int j = 0; j < ENC / 16; ++j) {      // 4 N tiles
    v16h bf  = load_b_w(sW, 2 * T_IN, j * 16, 0, lane);
    v8f acc0 = wmma16(aH[0], bf, v8f_zero());
    v8f acc1 = wmma16(aH[1], bf, v8f_zero());
    float mx = -3.4e38f;
#pragma unroll
    for (int r = 0; r < 8; ++r) mx = fmaxf(mx, fmaxf(acc0[r], acc1[r]));
    mx = fmaxf(mx, __shfl_xor(mx, 16, 32));  // combine the two row halves
    if (lane < 16) {
      int col = j * 16 + lane;
      scene[b * ENC + col] = lrelu(mx + bnbr[col]);
    }
  }
}

// ---------------------------------------------------------------------------
// K3: encoder LSTM.  One wave per 16 batch rows; recurrence in registers,
// h transposed D(f32)->A(f16) through a private LDS slab each step.
// Whh/Wih staged in LDS so the serial loop reads weights via ds_load.
// gates(16x256) = emb_t(16x32)@Wih^T + h(16x64)@Whh^T + (bih+bhh)
// ---------------------------------------------------------------------------
__global__ void encoder_lstm_kernel(const half_t* __restrict__ emb,
                                    const half_t* __restrict__ WihH /*[256,32]*/,
                                    const half_t* __restrict__ WhhH /*[256,64]*/,
                                    const float* __restrict__ bih,
                                    const float* __restrict__ bhh,
                                    half_t* __restrict__ h_out /*[B,ENC] f16*/) {
  __shared__ __align__(32) half_t sWih[4 * ENC * EMB];   // 16 KB
  __shared__ __align__(32) half_t sWhh[4 * ENC * ENC];   // 32 KB
  __shared__ __align__(32) half_t hbuf[4][16][ENC];      //  8 KB
  stage_lds(sWih, WihH, 4 * ENC * EMB, threadIdx.x, blockDim.x);
  stage_lds(sWhh, WhhH, 4 * ENC * ENC, threadIdx.x, blockDim.x);
  __syncthreads();

  int lane  = threadIdx.x & 31;
  int wv    = threadIdx.x >> 5;
  int b0    = (blockIdx.x * 4 + wv) * 16;
  int m     = lane & 15;
  int rbase = ((lane >> 4) & 1) << 3;

  // hoist the per-gate-column bias out of the recurrence
  float bias[16];
#pragma unroll
  for (int nt = 0; nt < 16; ++nt)
    bias[nt] = bih[nt * 16 + m] + bhh[nt * 16 + m];

  v16h h0 = v16h_zero(), h1 = v16h_zero();
  v8f  c[4];
#pragma unroll
  for (int j = 0; j < 4; ++j) c[j] = v8f_zero();

#pragma unroll 1
  for (int t = 0; t < T_IN; ++t) {
    v16h ax = load_a_f16(emb + (size_t)(t * BATCH + b0) * EMB, EMB, 0, lane);
#pragma unroll
    for (int j = 0; j < 4; ++j) {           // hidden tiles (cols 16j..16j+15)
      v8f gacc[4];
#pragma unroll
      for (int gi = 0; gi < 4; ++gi) {      // i,f,g,o
        int ntile = gi * 4 + j;
        v8f acc = v8f_zero();
        acc = wmma16(ax, load_b_w(sWih, EMB, ntile * 16, 0,  lane), acc);
        acc = wmma16(h0, load_b_w(sWhh, ENC, ntile * 16, 0,  lane), acc);
        acc = wmma16(h1, load_b_w(sWhh, ENC, ntile * 16, 32, lane), acc);
        float bb = bias[ntile];
#pragma unroll
        for (int r = 0; r < 8; ++r) acc[r] += bb;
        gacc[gi] = acc;
      }
#pragma unroll
      for (int r = 0; r < 8; ++r) {
        float iv = sigf(gacc[0][r]);
        float fv = sigf(gacc[1][r]);
        float gv = tanhf(gacc[2][r]);
        float ov = sigf(gacc[3][r]);
        float cn = fv * c[j][r] + iv * gv;
        c[j][r]  = cn;
        hbuf[wv][rbase + r][j * 16 + m] = (half_t)(ov * tanhf(cn));
      }
    }
    __syncthreads();
    h0 = load_a_f16(&hbuf[wv][0][0], ENC, 0,  lane);
    h1 = load_a_f16(&hbuf[wv][0][0], ENC, 32, lane);
    __syncthreads();
  }
  // final hidden state (still resident in LDS) -> global f16
  for (int i = lane; i < 16 * ENC; i += 32) {
    int rr = i >> 6, cc = i & (ENC - 1);
    h_out[(size_t)(b0 + rr) * ENC + cc] = hbuf[wv][rr][cc];
  }
}

// ---------------------------------------------------------------------------
// K4: enc = [ lrelu(h_enc@Wdyn^T+bdyn) , lrelu(scene@Wdyn^T+bdyn) ]  -> f16
// ---------------------------------------------------------------------------
__global__ void dyn_proj_kernel(const half_t* __restrict__ hEnc /*[B,64] f16*/,
                                const float* __restrict__ scene /*[B,64] f32*/,
                                const half_t* __restrict__ WdynH /*[32,64]*/,
                                const float* __restrict__ bdyn,
                                half_t* __restrict__ enc /*[B,64] f16*/) {
  int lane  = threadIdx.x & 31;
  int b0    = (blockIdx.x * 4 + (threadIdx.x >> 5)) * 16;
  int m     = lane & 15;
  int rbase = ((lane >> 4) & 1) << 3;

  v16h ah0 = load_a_f16(hEnc + (size_t)b0 * ENC, ENC, 0,  lane);
  v16h ah1 = load_a_f16(hEnc + (size_t)b0 * ENC, ENC, 32, lane);
  v16h as0, as1;
#pragma unroll
  for (int g = 0; g < 8; ++g) {
    int k = a_k_of_g(g, lane);
    const float* sp = scene + (size_t)(b0 + m) * ENC;
    as0[2 * g]     = (half_t)sp[k];
    as0[2 * g + 1] = (half_t)sp[k + 1];
    as1[2 * g]     = (half_t)sp[32 + k];
    as1[2 * g + 1] = (half_t)sp[32 + k + 1];
  }

#pragma unroll
  for (int src = 0; src < 2; ++src) {
    v16h a0 = src ? as0 : ah0;
    v16h a1 = src ? as1 : ah1;
#pragma unroll
    for (int j = 0; j < DYN / 16; ++j) {    // 2 N tiles
      v8f acc = v8f_zero();
      acc = wmma16(a0, load_b_w(WdynH, ENC, j * 16, 0,  lane), acc);
      acc = wmma16(a1, load_b_w(WdynH, ENC, j * 16, 32, lane), acc);
      float bb = bdyn[j * 16 + m];
#pragma unroll
      for (int r = 0; r < 8; ++r) {
        float v = lrelu(acc[r] + bb);
        enc[(size_t)(b0 + rbase + r) * (2 * DYN) + src * DYN + j * 16 + m] =
            (half_t)v;
      }
    }
  }
}

// ---------------------------------------------------------------------------
// K5: time-invariant decoder input projection (hoisted out of the 25-step
// loop):  xp = enc@Wih_d^T + (bih_d + bhh_d)   -> [B,512] f32
// ---------------------------------------------------------------------------
__global__ void dec_xproj_kernel(const half_t* __restrict__ enc /*[B,64]*/,
                                 const half_t* __restrict__ WihdH /*[512,64]*/,
                                 const float* __restrict__ bihd,
                                 const float* __restrict__ bhhd,
                                 float* __restrict__ xp /*[B,512]*/) {
  int lane  = threadIdx.x & 31;
  int b0    = (blockIdx.x * 4 + (threadIdx.x >> 5)) * 16;
  int m     = lane & 15;
  int rbase = ((lane >> 4) & 1) << 3;

  v16h a0 = load_a_f16(enc + (size_t)b0 * (2 * DYN), 2 * DYN, 0,  lane);
  v16h a1 = load_a_f16(enc + (size_t)b0 * (2 * DYN), 2 * DYN, 32, lane);

#pragma unroll 1
  for (int ntile = 0; ntile < (4 * DEC) / 16; ++ntile) {  // 32 tiles
    v8f acc = v8f_zero();
    acc = wmma16(a0, load_b_w(WihdH, 2 * DYN, ntile * 16, 0,  lane), acc);
    acc = wmma16(a1, load_b_w(WihdH, 2 * DYN, ntile * 16, 32, lane), acc);
    float bb = bihd[ntile * 16 + m] + bhhd[ntile * 16 + m];
#pragma unroll
    for (int r = 0; r < 8; ++r)
      xp[(size_t)(b0 + rbase + r) * (4 * DEC) + ntile * 16 + m] = acc[r] + bb;
  }
}

// ---------------------------------------------------------------------------
// K6: decoder LSTM (25 steps) + fused output head.  Whh_d (128 KB f16) is
// staged in LDS (320 KB/WGP on CDNA5) so every recurrent B-fragment read is
// a ds_load, keeping global latency off the serial dependency chain.
// gates(16x512) = xp + h(16x128)@Whh_d^T ; fut = h_t@Wop^T + bop every step.
// ---------------------------------------------------------------------------
__global__ void decoder_lstm_kernel(const float* __restrict__ xp /*[B,512]*/,
                                    const half_t* __restrict__ WhhdH /*[512,128]*/,
                                    const float* __restrict__ Wop /*[2,128]*/,
                                    const float* __restrict__ bop,
                                    float* __restrict__ fut /*[T_OUT,B,2]*/) {
  __shared__ __align__(32) half_t sWhh[4 * DEC * DEC];   // 128 KB
  __shared__ __align__(32) half_t hbuf[4][16][DEC];      //  16 KB
  stage_lds(sWhh, WhhdH, 4 * DEC * DEC, threadIdx.x, blockDim.x);
  __syncthreads();

  int lane  = threadIdx.x & 31;
  int wv    = threadIdx.x >> 5;
  int b0    = (blockIdx.x * 4 + wv) * 16;
  int m     = lane & 15;
  int rbase = ((lane >> 4) & 1) << 3;

  // hoist output-head weights/bias out of the recurrence
  float w0a[8], w1a[8];
#pragma unroll
  for (int j = 0; j < 8; ++j) {
    w0a[j] = Wop[0 * DEC + j * 16 + m];
    w1a[j] = Wop[1 * DEC + j * 16 + m];
  }
  float bop0 = bop[0], bop1 = bop[1];

  v16h h[4];
#pragma unroll
  for (int kc = 0; kc < 4; ++kc) h[kc] = v16h_zero();
  v8f c[8];
#pragma unroll
  for (int j = 0; j < 8; ++j) c[j] = v8f_zero();

#pragma unroll 1
  for (int t = 0; t < T_OUT; ++t) {
    float f0[8], f1[8];
#pragma unroll
    for (int r = 0; r < 8; ++r) { f0[r] = 0.f; f1[r] = 0.f; }

#pragma unroll
    for (int j = 0; j < DEC / 16; ++j) {    // 8 hidden tiles
      v8f gacc[4];
#pragma unroll
      for (int gi = 0; gi < 4; ++gi) {      // i,f,g,o
        int ntile = gi * 8 + j;
        v8f acc;
#pragma unroll
        for (int r = 0; r < 8; ++r)
          acc[r] = xp[(size_t)(b0 + rbase + r) * (4 * DEC) + ntile * 16 + m];
#pragma unroll
        for (int kc = 0; kc < 4; ++kc)
          acc = wmma16(h[kc], load_b_w(sWhh, DEC, ntile * 16, kc * 32, lane),
                       acc);
        gacc[gi] = acc;
      }
#pragma unroll
      for (int r = 0; r < 8; ++r) {
        float iv = sigf(gacc[0][r]);
        float fv = sigf(gacc[1][r]);
        float gv = tanhf(gacc[2][r]);
        float ov = sigf(gacc[3][r]);
        float cn = fv * c[j][r] + iv * gv;
        c[j][r]  = cn;
        float hn = ov * tanhf(cn);
        hbuf[wv][rbase + r][j * 16 + m] = (half_t)hn;
        f0[r] += hn * w0a[j];               // fused output head partials
        f1[r] += hn * w1a[j];
      }
    }
    __syncthreads();
#pragma unroll
    for (int kc = 0; kc < 4; ++kc)
      h[kc] = load_a_f16(&hbuf[wv][0][0], DEC, kc * 32, lane);
    __syncthreads();

    // reduce fut over the 16 lanes of each row-half
#pragma unroll
    for (int r = 0; r < 8; ++r) {
#pragma unroll
      for (int mask = 1; mask < 16; mask <<= 1) {
        f0[r] += __shfl_xor(f0[r], mask, 32);
        f1[r] += __shfl_xor(f1[r], mask, 32);
      }
    }
    if (m == 0) {                            // lanes 0 and 16
#pragma unroll
      for (int r = 0; r < 8; ++r) {
        int row = b0 + rbase + r;
        fut[((size_t)t * BATCH + row) * 2 + 0] = f0[r] + bop0;
        fut[((size_t)t * BATCH + row) * 2 + 1] = f1[r] + bop1;
      }
    }
  }
}

// ---------------------------------------------------------------------------
// Host launcher
// ---------------------------------------------------------------------------
extern "C" void kernel_launch(void* const* d_in, const int* in_sizes, int n_in,
                              void* d_out, int out_size, void* d_ws,
                              size_t ws_size, hipStream_t stream) {
  const float* hist  = (const float*)d_in[0];
  const float* nbrs  = (const float*)d_in[1];
  const float* Wip   = (const float*)d_in[2];
  const float* bip   = (const float*)d_in[3];
  const float* Wih_e = (const float*)d_in[4];
  const float* Whh_e = (const float*)d_in[5];
  const float* bih_e = (const float*)d_in[6];
  const float* bhh_e = (const float*)d_in[7];
  const float* Wdyn  = (const float*)d_in[8];
  const float* bdyn  = (const float*)d_in[9];
  const float* Wnbr  = (const float*)d_in[10];
  const float* bnbr  = (const float*)d_in[11];
  const float* Wih_d = (const float*)d_in[12];
  const float* Whh_d = (const float*)d_in[13];
  const float* bih_d = (const float*)d_in[14];
  const float* bhh_d = (const float*)d_in[15];
  const float* Wop   = (const float*)d_in[16];
  const float* bop   = (const float*)d_in[17];
  // d_in[18] (segment_ids) is implied by layout: seg[n] = n >> 5.
  float* fut = (float*)d_out;

  // --- workspace carve-up (all offsets 4 KB aligned) ---
  char* ws = (char*)d_ws;
  size_t off = 0;
  auto carve = [&](size_t bytes) {
    char* p = ws + off;
    off = (off + bytes + 4095) & ~(size_t)4095;
    return p;
  };
  half_t* WihE_h = (half_t*)carve(4 * ENC * EMB * sizeof(half_t));       // [256,32]
  half_t* WhhE_h = (half_t*)carve(4 * ENC * ENC * sizeof(half_t));       // [256,64]
  half_t* Wnbr_h = (half_t*)carve(ENC * 2 * T_IN * sizeof(half_t));      // [64,32]
  half_t* Wdyn_h = (half_t*)carve(DYN * ENC * sizeof(half_t));           // [32,64]
  half_t* Wihd_h = (half_t*)carve(4 * DEC * 2 * DYN * sizeof(half_t));   // [512,64]
  half_t* Whhd_h = (half_t*)carve(4 * DEC * DEC * sizeof(half_t));       // [512,128]
  half_t* emb_h  = (half_t*)carve((size_t)T_IN * BATCH * EMB * sizeof(half_t));
  float*  scene  = (float*)carve((size_t)BATCH * ENC * sizeof(float));
  half_t* hEnc_h = (half_t*)carve((size_t)BATCH * ENC * sizeof(half_t));
  half_t* enc_h  = (half_t*)carve((size_t)BATCH * 2 * DYN * sizeof(half_t));
  float*  xp     = (float*)carve((size_t)BATCH * 4 * DEC * sizeof(float));
  (void)ws_size; (void)n_in; (void)in_sizes; (void)out_size;

  // K0: weight conversions to f16
  auto cvt = [&](const float* s, half_t* d, int n) {
    f32_to_f16_kernel<<<(n + 255) / 256, 256, 0, stream>>>(s, d, n);
  };
  cvt(Wih_e, WihE_h, 4 * ENC * EMB);
  cvt(Whh_e, WhhE_h, 4 * ENC * ENC);
  cvt(Wnbr,  Wnbr_h, ENC * 2 * T_IN);
  cvt(Wdyn,  Wdyn_h, DYN * ENC);
  cvt(Wih_d, Wihd_h, 4 * DEC * 2 * DYN);
  cvt(Whh_d, Whhd_h, 4 * DEC * DEC);

  // K1: ego embedding
  embed_kernel<<<(T_IN * BATCH * EMB) / 256, 256, 0, stream>>>(hist, Wip, bip,
                                                               emb_h);
  // K2: neighbor MLP + segment max  (1 wave / hero, 4 waves / block)
  neighbor_scene_kernel<<<BATCH / 4, 128, 0, stream>>>(hist, nbrs, Wnbr_h,
                                                       bnbr, scene);
  // K3: encoder LSTM (1 wave / 16 rows -> 256 waves)
  encoder_lstm_kernel<<<BATCH / 64, 128, 0, stream>>>(emb_h, WihE_h, WhhE_h,
                                                      bih_e, bhh_e, hEnc_h);
  // K4: dyn projections + concat
  dyn_proj_kernel<<<BATCH / 64, 128, 0, stream>>>(hEnc_h, scene, Wdyn_h, bdyn,
                                                  enc_h);
  // K5: hoisted decoder input projection
  dec_xproj_kernel<<<BATCH / 64, 128, 0, stream>>>(enc_h, Wihd_h, bih_d,
                                                   bhh_d, xp);
  // K6: decoder LSTM + fused output head
  decoder_lstm_kernel<<<BATCH / 64, 128, 0, stream>>>(xp, Whhd_h, Wop, bop,
                                                      fut);
}
